// MultiHeadRelativeAttention_9113920602584
// MI455X (gfx1250) — matrix operand order repaired and checked by reading, not compiled, not run on hardware
//
#include <hip/hip_runtime.h>
#include <hip/hip_bf16.h>
#include <math.h>

typedef __bf16 bf16_t;
typedef __attribute__((ext_vector_type(16))) __bf16 bf16x16;
typedef __attribute__((ext_vector_type(8)))  __bf16 bf16x8;
typedef __attribute__((ext_vector_type(8)))  float  f32x8;

#define S_LEN   2048
#define BATCH   2
#define NHEAD   8
#define DMODEL  512
#define HDIM    64
#define NRPR    513
#define RPAD    576   /* 513 padded to 18*32 so K-loops and 64-wide N tiles divide evenly */

// ---------------- WMMA fragment helpers (wave32, 16x16x32 bf16) ----------------

// A fragment 16x32: lane<16 holds row (lane), K = k0..k0+15
//                   lane>=16 holds row (lane-16), K = k0+16..31
static __device__ __forceinline__ bf16x16 frag_a(const bf16_t* base, int r0, int k0,
                                                 int ld, int lane) {
  const bf16_t* p = base + (size_t)(r0 + (lane & 15)) * ld + k0 + (lane & 16);
  bf16x8 lo = *(const bf16x8*)p;
  bf16x8 hi = *(const bf16x8*)(p + 8);
  bf16x16 r;
#pragma unroll
  for (int t = 0; t < 8; ++t) { r[t] = lo[t]; r[t + 8] = hi[t]; }
  return r;
}

// B fragment 32x16 from a transposed (N-major) operand BT[n][k]:
// lane<16 holds column n0+lane, K = k0..k0+15 ; lane>=16 column n0+lane-16, K = k0+16..31
static __device__ __forceinline__ bf16x16 frag_b(const bf16_t* baseT, int n0, int k0,
                                                 int ld, int lane) {
  const bf16_t* p = baseT + (size_t)(n0 + (lane & 15)) * ld + k0 + (lane & 16);
  return *(const bf16x16*)p;
}

static __device__ __forceinline__ f32x8 wmma_bf16(bf16x16 a, bf16x16 b, f32x8 c) {
  return __builtin_amdgcn_wmma_f32_16x16x32_bf16(false, a, false, b, (short)0, c,
                                                 false, false);
}

// ---------------- Prep kernels ----------------

__global__ void k_cvt(const float* __restrict__ in, bf16_t* __restrict__ out, int n) {
  int t = blockIdx.x * blockDim.x + threadIdx.x;
  if (t < n) out[t] = (bf16_t)in[t];
}

// WT[o][i] = W[i][o], 512x512, bf16
__global__ void k_tcvt(const float* __restrict__ W, bf16_t* __restrict__ WT) {
  int t = blockIdx.x * blockDim.x + threadIdx.x;
  int o = t >> 9, i = t & 511;
  WT[(size_t)o * DMODEL + i] = (bf16_t)W[(size_t)i * DMODEL + o];
}

// Ek16[r][d] (row-major, r zero-padded to RPAD) and EvT16[d][r] (transposed, padded)
__global__ void k_ekprep(const float* __restrict__ Ek, const float* __restrict__ Ev,
                         bf16_t* __restrict__ Ek16, bf16_t* __restrict__ EvT16) {
  int t = blockIdx.x * blockDim.x + threadIdx.x;   // RPAD*64 threads
  int r = t >> 6, d = t & 63;
  float ek = (r < NRPR) ? Ek[(size_t)r * HDIM + d] : 0.0f;
  float ev = (r < NRPR) ? Ev[(size_t)r * HDIM + d] : 0.0f;
  Ek16[(size_t)r * HDIM + d] = (bf16_t)ek;
  EvT16[(size_t)d * RPAD + r] = (bf16_t)ev;
}

// ---------------- Generic GEMM, bf16 out: C[M x N] = A[M x K] * BT[N x K]^T ----------------
// One wave per 16(M) x 64(N) strip; blockDim = 256 (8 waves).
__global__ void k_gemm_bf16(const bf16_t* __restrict__ A, const bf16_t* __restrict__ BT,
                            bf16_t* __restrict__ C, int N64, int K,
                            int lda, int ldb, int ldc) {
  int wave = (blockIdx.x * blockDim.x + threadIdx.x) >> 5;
  int lane = threadIdx.x & 31;
  int mt = wave / N64, nt = wave % N64;
  int m0 = mt * 16, n0 = nt * 64;
  f32x8 acc[4] = {};
  for (int k0 = 0; k0 < K; k0 += 32) {
    bf16x16 a = frag_a(A, m0, k0, lda, lane);
#pragma unroll
    for (int t = 0; t < 4; ++t)
      acc[t] = wmma_bf16(a, frag_b(BT, n0 + 16 * t, k0, ldb, lane), acc[t]);
  }
  int n = lane & 15, mo = (lane >> 4) << 3;
#pragma unroll
  for (int r = 0; r < 8; ++r) {
    bf16_t* row = C + (size_t)(m0 + mo + r) * ldc + n0 + n;
#pragma unroll
    for (int t = 0; t < 4; ++t) row[16 * t] = (bf16_t)acc[t][r];
  }
}

// V projection with transposed store: VT16[((b*H+h)*64+d)*S + s]
__global__ void k_vproj(const bf16_t* __restrict__ val16, const bf16_t* __restrict__ WvT,
                        bf16_t* __restrict__ VT16) {
  int wave = (blockIdx.x * blockDim.x + threadIdx.x) >> 5;
  int lane = threadIdx.x & 31;
  int mt = wave / 8, nt = wave % 8;           // M=4096 -> 256 mtiles, N=512 -> 8 n64
  int m0 = mt * 16, n0 = nt * 64;
  f32x8 acc[4] = {};
  for (int k0 = 0; k0 < DMODEL; k0 += 32) {
    bf16x16 a = frag_a(val16, m0, k0, DMODEL, lane);
#pragma unroll
    for (int t = 0; t < 4; ++t)
      acc[t] = wmma_bf16(a, frag_b(WvT, n0 + 16 * t, k0, DMODEL, lane), acc[t]);
  }
  int n = lane & 15, mo = (lane >> 4) << 3;
#pragma unroll
  for (int r = 0; r < 8; ++r) {
    int tr = m0 + mo + r;
    int b = tr >> 11, s = tr & (S_LEN - 1);
#pragma unroll
    for (int t = 0; t < 4; ++t) {
      int col = n0 + 16 * t + n;
      int h = col >> 6, d = col & 63;
      VT16[(((size_t)b * NHEAD + h) * HDIM + d) * S_LEN + s] = (bf16_t)acc[t][r];
    }
  }
}

// QE16[bh][i][r] = q(b,h) @ Ek^T, padded N=RPAD. grid: x=144 blocks, y=bh
__global__ void k_qe(const bf16_t* __restrict__ Q16, const bf16_t* __restrict__ Ek16,
                     bf16_t* __restrict__ QE16) {
  int wave = (blockIdx.x * blockDim.x + threadIdx.x) >> 5;  // 1152 waves/bh
  int lane = threadIdx.x & 31;
  int bh = blockIdx.y;
  int b = bh >> 3, h = bh & 7;
  int mt = wave / (RPAD / 64), nt = wave % (RPAD / 64);
  int m0 = mt * 16, n0 = nt * 64;
  const bf16_t* A = Q16 + ((size_t)b * S_LEN) * DMODEL + h * HDIM;
  bf16_t* C = QE16 + (size_t)bh * S_LEN * RPAD;
  f32x8 acc[4] = {};
  for (int k0 = 0; k0 < HDIM; k0 += 32) {
    bf16x16 a = frag_a(A, m0, k0, DMODEL, lane);
#pragma unroll
    for (int t = 0; t < 4; ++t)
      acc[t] = wmma_bf16(a, frag_b(Ek16, n0 + 16 * t, k0, HDIM, lane), acc[t]);
  }
  int n = lane & 15, mo = (lane >> 4) << 3;
#pragma unroll
  for (int r = 0; r < 8; ++r) {
    bf16_t* row = C + (size_t)(m0 + mo + r) * RPAD + n0 + n;
#pragma unroll
    for (int t = 0; t < 4; ++t) row[16 * t] = (bf16_t)acc[t][r];
  }
}

// alpha = (Q K^T + gather(QE)) * 0.125, fp32 into d_out attn region. grid x=512, y=bh
__global__ void k_alpha(const bf16_t* __restrict__ Q16, const bf16_t* __restrict__ K16,
                        const bf16_t* __restrict__ QE16, float* __restrict__ attnF) {
  int wave = (blockIdx.x * blockDim.x + threadIdx.x) >> 5;  // 4096 waves/bh
  int lane = threadIdx.x & 31;
  int bh = blockIdx.y;
  int b = bh >> 3, h = bh & 7;
  int mt = wave / (S_LEN / 64), nt = wave % (S_LEN / 64);
  int i0 = mt * 16, j0 = nt * 64;
  const bf16_t* A  = Q16 + ((size_t)b * S_LEN) * DMODEL + h * HDIM;
  const bf16_t* BT = K16 + ((size_t)b * S_LEN) * DMODEL + h * HDIM;   // rows j, K=d
  f32x8 acc[4] = {};
  for (int k0 = 0; k0 < HDIM; k0 += 32) {
    bf16x16 a = frag_a(A, i0, k0, DMODEL, lane);
#pragma unroll
    for (int t = 0; t < 4; ++t)
      acc[t] = wmma_bf16(a, frag_b(BT, j0 + 16 * t, k0, DMODEL, lane), acc[t]);
  }
  const bf16_t* qe = QE16 + (size_t)bh * S_LEN * RPAD;
  float* outp = attnF + (size_t)bh * S_LEN * S_LEN;
  int n = lane & 15, mo = (lane >> 4) << 3;
#pragma unroll
  for (int r = 0; r < 8; ++r) {
    int i = i0 + mo + r;
    const bf16_t* qrow = qe + (size_t)i * RPAD;
    float* orow = outp + (size_t)i * S_LEN;
#pragma unroll
    for (int t = 0; t < 4; ++t) {
      int j = j0 + 16 * t + n;
      int rel = j - i;
      rel = rel < -256 ? -256 : (rel > 256 ? 256 : rel);
      orow[j] = (acc[t][r] + (float)qrow[rel + 256]) * 0.125f;
    }
  }
}

// Row softmax (2048 wide): normalize fp32 in place (d_out) and write bf16 copy.
__global__ void k_softmax(float* __restrict__ attnF, bf16_t* __restrict__ attn16) {
  size_t row = blockIdx.x;
  float*  p   = attnF  + row * S_LEN;
  bf16_t* p16 = attn16 + row * S_LEN;
  __shared__ float red[256];
  int tid = threadIdx.x;
  float v[8];
  float mx = -3.0e38f;
#pragma unroll
  for (int t = 0; t < 8; ++t) { v[t] = p[tid + t * 256]; mx = fmaxf(mx, v[t]); }
  red[tid] = mx; __syncthreads();
  for (int s = 128; s > 0; s >>= 1) {
    if (tid < s) red[tid] = fmaxf(red[tid], red[tid + s]);
    __syncthreads();
  }
  mx = red[0]; __syncthreads();
  float sum = 0.0f;
#pragma unroll
  for (int t = 0; t < 8; ++t) { v[t] = __expf(v[t] - mx); sum += v[t]; }
  red[tid] = sum; __syncthreads();
  for (int s = 128; s > 0; s >>= 1) {
    if (tid < s) red[tid] += red[tid + s];
    __syncthreads();
  }
  float inv = 1.0f / red[0];
#pragma unroll
  for (int t = 0; t < 8; ++t) {
    float w = v[t] * inv;
    p[tid + t * 256]   = w;
    p16[tid + t * 256] = (bf16_t)w;
  }
}

// Relative-bucket aggregation: P[i][r] = sum_{j: clip(j-i)+256 == r} attn[i][j]
__global__ void k_pbucket(const float* __restrict__ attnF, bf16_t* __restrict__ P16) {
  size_t row = blockIdx.x;                 // bh*S + i
  int i = (int)(row & (S_LEN - 1));
  const float* p = attnF + row * S_LEN;
  bf16_t* out = P16 + row * RPAD;
  __shared__ float red[256];
  __shared__ float s0s, s1s;
  int tid = threadIdx.x;
  float s = 0.0f;
  for (int j = tid; j <= i - 256; j += 256) s += p[j];       // bucket 0: j-i <= -256
  red[tid] = s; __syncthreads();
  for (int k = 128; k > 0; k >>= 1) { if (tid < k) red[tid] += red[tid + k]; __syncthreads(); }
  if (tid == 0) s0s = red[0];
  __syncthreads();
  s = 0.0f;
  for (int j = i + 256 + tid; j < S_LEN; j += 256) s += p[j]; // bucket 512: j-i >= 256
  red[tid] = s; __syncthreads();
  for (int k = 128; k > 0; k >>= 1) { if (tid < k) red[tid] += red[tid + k]; __syncthreads(); }
  if (tid == 0) s1s = red[0];
  __syncthreads();
  for (int r = tid; r < RPAD; r += 256) {
    float val;
    if (r == 0) val = s0s;
    else if (r == 512) val = s1s;
    else if (r < 512) {
      int j = i + r - 256;
      val = (j >= 0 && j < S_LEN) ? p[j] : 0.0f;
    } else val = 0.0f;
    out[r] = (bf16_t)val;
  }
}

// scores = attn16 @ V + P16 @ Ev -> concat16[b][s][h*64+d]. grid x=16 blocks, y=bh
__global__ void k_scores(const bf16_t* __restrict__ attn16, const bf16_t* __restrict__ VT16,
                         const bf16_t* __restrict__ P16, const bf16_t* __restrict__ EvT16,
                         bf16_t* __restrict__ concat16) {
  int wave = (blockIdx.x * blockDim.x + threadIdx.x) >> 5;   // 128 waves/bh, N=64 fits one strip
  int lane = threadIdx.x & 31;
  int bh = blockIdx.y;
  int b = bh >> 3, h = bh & 7;
  int m0 = wave * 16;
  const bf16_t* A1 = attn16 + (size_t)bh * S_LEN * S_LEN;    // (S x S)
  const bf16_t* B1 = VT16   + (size_t)bh * HDIM * S_LEN;     // rows d, K=j
  f32x8 acc[4] = {};
  for (int k0 = 0; k0 < S_LEN; k0 += 32) {
    bf16x16 a = frag_a(A1, m0, k0, S_LEN, lane);
#pragma unroll
    for (int t = 0; t < 4; ++t)
      acc[t] = wmma_bf16(a, frag_b(B1, 16 * t, k0, S_LEN, lane), acc[t]);
  }
  const bf16_t* A2 = P16 + (size_t)bh * S_LEN * RPAD;        // (S x RPAD)
  for (int k0 = 0; k0 < RPAD; k0 += 32) {
    bf16x16 a = frag_a(A2, m0, k0, RPAD, lane);
#pragma unroll
    for (int t = 0; t < 4; ++t)
      acc[t] = wmma_bf16(a, frag_b(EvT16, 16 * t, k0, RPAD, lane), acc[t]);
  }
  int n = lane & 15, mo = (lane >> 4) << 3;
#pragma unroll
  for (int r = 0; r < 8; ++r) {
    int i = m0 + mo + r;
    bf16_t* row = concat16 + ((size_t)b * S_LEN + i) * DMODEL + h * HDIM + n;
#pragma unroll
    for (int t = 0; t < 4; ++t) row[16 * t] = (bf16_t)acc[t][r];
  }
}

// out = concat16 @ Wo + bo (fp32 to d_out)
__global__ void k_out(const bf16_t* __restrict__ concat16, const bf16_t* __restrict__ WoT,
                      const float* __restrict__ bo, float* __restrict__ out) {
  int wave = (blockIdx.x * blockDim.x + threadIdx.x) >> 5;
  int lane = threadIdx.x & 31;
  int mt = wave / 8, nt = wave % 8;
  int m0 = mt * 16, n0 = nt * 64;
  f32x8 acc[4] = {};
  for (int k0 = 0; k0 < DMODEL; k0 += 32) {
    bf16x16 a = frag_a(concat16, m0, k0, DMODEL, lane);
#pragma unroll
    for (int t = 0; t < 4; ++t)
      acc[t] = wmma_bf16(a, frag_b(WoT, n0 + 16 * t, k0, DMODEL, lane), acc[t]);
  }
  int n = lane & 15, mo = (lane >> 4) << 3;
#pragma unroll
  for (int r = 0; r < 8; ++r) {
    float* row = out + (size_t)(m0 + mo + r) * DMODEL + n0 + n;
#pragma unroll
    for (int t = 0; t < 4; ++t) {
      int col = n0 + 16 * t + n;
      row[16 * t] = acc[t][r] + bo[col];
    }
  }
}

// ---------------- Launch ----------------

extern "C" void kernel_launch(void* const* d_in, const int* in_sizes, int n_in,
                              void* d_out, int out_size, void* d_ws, size_t ws_size,
                              hipStream_t stream) {
  const float* query = (const float*)d_in[0];
  const float* value = (const float*)d_in[1];
  const float* Wq = (const float*)d_in[2];
  const float* Wk = (const float*)d_in[3];
  const float* Wv = (const float*)d_in[4];
  const float* Wo = (const float*)d_in[5];
  const float* bo = (const float*)d_in[6];
  const float* Ek = (const float*)d_in[7];
  const float* Ev = (const float*)d_in[8];

  float* outF  = (float*)d_out;                             // (B,S,512)
  float* attnF = outF + (size_t)BATCH * S_LEN * DMODEL;     // (B,H,S,S) fp32

  size_t off = 0;
  auto wsp = [&](size_t bytes) -> void* {
    void* p = (char*)d_ws + off;
    off += (bytes + 255) & ~(size_t)255;
    return p;
  };
  const size_t NROWS = (size_t)BATCH * S_LEN;               // 4096
  bf16_t* qry16  = (bf16_t*)wsp(NROWS * DMODEL * 2);
  bf16_t* val16  = (bf16_t*)wsp(NROWS * DMODEL * 2);
  bf16_t* WqT16  = (bf16_t*)wsp((size_t)DMODEL * DMODEL * 2);
  bf16_t* WkT16  = (bf16_t*)wsp((size_t)DMODEL * DMODEL * 2);
  bf16_t* WvT16  = (bf16_t*)wsp((size_t)DMODEL * DMODEL * 2);
  bf16_t* WoT16  = (bf16_t*)wsp((size_t)DMODEL * DMODEL * 2);
  bf16_t* Ek16   = (bf16_t*)wsp((size_t)RPAD * HDIM * 2);
  bf16_t* EvT16  = (bf16_t*)wsp((size_t)HDIM * RPAD * 2);
  bf16_t* Q16    = (bf16_t*)wsp(NROWS * DMODEL * 2);
  bf16_t* K16    = (bf16_t*)wsp(NROWS * DMODEL * 2);
  bf16_t* VT16   = (bf16_t*)wsp(NROWS * DMODEL * 2);
  bf16_t* QE16   = (bf16_t*)wsp((size_t)BATCH * NHEAD * S_LEN * RPAD * 2);
  bf16_t* attn16 = (bf16_t*)wsp((size_t)BATCH * NHEAD * S_LEN * S_LEN * 2);
  bf16_t* P16    = (bf16_t*)wsp((size_t)BATCH * NHEAD * S_LEN * RPAD * 2);
  bf16_t* cat16  = (bf16_t*)wsp(NROWS * DMODEL * 2);
  (void)ws_size; (void)in_sizes; (void)n_in; (void)out_size;

  int nElem = (int)(NROWS * DMODEL);                        // 2,097,152
  k_cvt<<<nElem / 256, 256, 0, stream>>>(query, qry16, nElem);
  k_cvt<<<nElem / 256, 256, 0, stream>>>(value, val16, nElem);
  k_tcvt<<<(DMODEL * DMODEL) / 256, 256, 0, stream>>>(Wq, WqT16);
  k_tcvt<<<(DMODEL * DMODEL) / 256, 256, 0, stream>>>(Wk, WkT16);
  k_tcvt<<<(DMODEL * DMODEL) / 256, 256, 0, stream>>>(Wv, WvT16);
  k_tcvt<<<(DMODEL * DMODEL) / 256, 256, 0, stream>>>(Wo, WoT16);
  k_ekprep<<<(RPAD * HDIM) / 256, 256, 0, stream>>>(Ek, Ev, Ek16, EvT16);

  // Projections: 4096x512x512, 2048 waves -> 256 blocks
  k_gemm_bf16<<<256, 256, 0, stream>>>(qry16, WqT16, Q16, 8, DMODEL, DMODEL, DMODEL, DMODEL);
  k_gemm_bf16<<<256, 256, 0, stream>>>(val16, WkT16, K16, 8, DMODEL, DMODEL, DMODEL, DMODEL);
  k_vproj<<<256, 256, 0, stream>>>(val16, WvT16, VT16);

  k_qe<<<dim3(144, BATCH * NHEAD), 256, 0, stream>>>(Q16, Ek16, QE16);
  k_alpha<<<dim3(512, BATCH * NHEAD), 256, 0, stream>>>(Q16, K16, QE16, attnF);
  k_softmax<<<BATCH * NHEAD * S_LEN, 256, 0, stream>>>(attnF, attn16);
  k_pbucket<<<BATCH * NHEAD * S_LEN, 256, 0, stream>>>(attnF, P16);
  k_scores<<<dim3(16, BATCH * NHEAD), 256, 0, stream>>>(attn16, VT16, P16, EvT16, cat16);
  k_out<<<256, 256, 0, stream>>>(cat16, WoT16, bo, outF);
}
